// MLPDecoder_57114475102416
// MI455X (gfx1250) — matrix operand order repaired and checked by reading, not compiled, not run on hardware
//
#include <hip/hip_runtime.h>
#include <hip/hip_bf16.h>

typedef __attribute__((ext_vector_type(16))) _Float16 v16h;
typedef __attribute__((ext_vector_type(8)))  _Float16 v8h;
typedef __attribute__((ext_vector_type(4)))  _Float16 v4h;
typedef __attribute__((ext_vector_type(8)))  float    v8f;

#define LATENT      64
#define KDIM        128          // 2*LATENT
#define HIDDEN      64
#define TILE_EDGES  128          // edges per block
#define ROW_HALFS   136          // 128 halfs + 8 pad halfs (16B) to spread LDS banks
#define BLOCK       256          // 8 waves of 32

// logits[e] = relu( concat(z[src],z[dst]) @ W1 + b1 ) @ W2 + b2
__global__ __launch_bounds__(BLOCK)
void mlp_decoder_wmma(const float* __restrict__ z,
                      const long long* __restrict__ ei,   // (2,E) int64
                      const float* __restrict__ W1,       // (128,64) row-major
                      const float* __restrict__ b1,       // (64)
                      const float* __restrict__ W2,       // (64)
                      const float* __restrict__ b2,       // (1)
                      float* __restrict__ out,            // (E)
                      int E)
{
    __shared__ _Float16 feat[TILE_EDGES * ROW_HALFS];   // 34816 B, f16 features
    __shared__ _Float16 w1t [HIDDEN    * ROW_HALFS];    // 17408 B, W1 transposed [n][k]
    __shared__ int s_src[TILE_EDGES];
    __shared__ int s_dst[TILE_EDGES];

    const int tid  = threadIdx.x;
    const int lane = tid & 31;
    const int wave = tid >> 5;
    const int eb   = blockIdx.x * TILE_EDGES;

    // ---- Stage W1 transposed into LDS as f16: w1t[n*ROW_HALFS + k] ----
    #pragma unroll
    for (int idx = tid; idx < KDIM * HIDDEN; idx += BLOCK) {
        const int k = idx >> 6;          // 0..127
        const int n = idx & 63;          // 0..63
        w1t[n * ROW_HALFS + k] = (_Float16)W1[idx];   // coalesced global read
    }

    // ---- Stage edge indices ----
    for (int i = tid; i < TILE_EDGES; i += BLOCK) {
        const int e = eb + i;
        const int ce = (e < E) ? e : 0;                // clamp (no store later)
        s_src[i] = (int)ei[ce];
        s_dst[i] = (int)ei[(size_t)E + ce];
    }
    __syncthreads();

    // ---- Gather features: feat[e][0:64] = z[src] (f16), feat[e][64:128] = z[dst] ----
    // 4096 float4 units: 32 per edge (16 src + 16 dst); 16 lanes stream a
    // contiguous 256-B z row -> fully coalesced per gathered row.
    #pragma unroll
    for (int u = tid; u < TILE_EDGES * 32; u += BLOCK) {
        const int el  = u >> 5;
        const int q   = u & 31;
        const int f4i = q & 15;
        const int row = (q < 16) ? s_src[el] : s_dst[el];
        const float4 v = ((const float4*)(z + (size_t)row * LATENT))[f4i];
        v4h h; h.x = (_Float16)v.x; h.y = (_Float16)v.y;
               h.z = (_Float16)v.z; h.w = (_Float16)v.w;
        *(v4h*)&feat[el * ROW_HALFS + ((q & 16) ? 64 : 0) + f4i * 4] = h;
    }
    __syncthreads();

    // ---- WMMA: each wave does a 16x64x128 GEMM tile = 16 x v_wmma_f32_16x16x32_f16 ----
    // A 16-bit 16x32 layout (ISA 7.12.2): lanes 0-15 hold K=kb+0..7 , kb+16..23
    //                                     lanes16-31 hold K=kb+8..15, kb+24..31
    const int hi   = (lane & 16) ? 1 : 0;
    const _Float16* frow = &feat[(wave * 16 + (lane & 15)) * ROW_HALFS];

    v16h a[4];
    #pragma unroll
    for (int kk = 0; kk < 4; ++kk) {
        const int kb = kk * 32 + hi * 8;
        const v8h lo = *(const v8h*)(frow + kb);
        const v8h hh = *(const v8h*)(frow + kb + 16);
        #pragma unroll
        for (int i = 0; i < 8; ++i) { a[kk][i] = lo[i]; a[kk][i + 8] = hh[i]; }
    }

    float pacc[8];
    #pragma unroll
    for (int r = 0; r < 8; ++r) pacc[r] = 0.0f;

    #pragma unroll
    for (int nn = 0; nn < 4; ++nn) {
        const int n = nn * 16 + (lane & 15);
        // B 32x16: N = lane%16; lanes<16: K 0..7, 16..23; lanes>=16: K 8..15, 24..31
        const _Float16* brow = &w1t[n * ROW_HALFS + hi * 8];

        v8f c = {};
        #pragma unroll
        for (int kk = 0; kk < 4; ++kk) {
            const v8h blo = *(const v8h*)(brow + kk * 32);
            const v8h bhi = *(const v8h*)(brow + kk * 32 + 16);
            v16h b;
            #pragma unroll
            for (int i = 0; i < 8; ++i) { b[i] = blo[i]; b[i + 8] = bhi[i]; }
            c = __builtin_amdgcn_wmma_f32_16x16x32_f16(
                    false, a[kk], false, b, (short)0, c, false, false);
        }

        // Fused layer 2: relu(c + b1[n]) * W2[n], accumulate per-lane per-row
        const float b1n = b1[n];
        const float w2n = W2[n];
        #pragma unroll
        for (int r = 0; r < 8; ++r) {
            float h = c[r] + b1n;
            h = (h > 0.0f) ? h : 0.0f;
            pacc[r] += h * w2n;
        }
    }

    // Butterfly-reduce over each 16-lane half (rows M=r in lanes 0-15, M=r+8 in 16-31)
    #pragma unroll
    for (int r = 0; r < 8; ++r) {
        float v = pacc[r];
        #pragma unroll
        for (int m = 1; m < 16; m <<= 1) v += __shfl_xor(v, m, 32);
        pacc[r] = v;
    }

    if ((lane & 15) == 0) {
        const float b2v = b2[0];
        const int mbase = (lane >> 4) * 8;          // lane 0 -> rows 0..7, lane 16 -> 8..15
        #pragma unroll
        for (int r = 0; r < 8; ++r) {
            const int e = eb + wave * 16 + mbase + r;
            if (e < E) out[e] = pacc[r] + b2v;
        }
    }
}

extern "C" void kernel_launch(void* const* d_in, const int* in_sizes, int n_in,
                              void* d_out, int out_size, void* d_ws, size_t ws_size,
                              hipStream_t stream) {
    const float*     z  = (const float*)d_in[0];
    const long long* ei = (const long long*)d_in[1];   // (2,E) int64
    const float*     W1 = (const float*)d_in[2];
    const float*     b1 = (const float*)d_in[3];
    const float*     W2 = (const float*)d_in[4];
    const float*     b2 = (const float*)d_in[5];
    float*           out = (float*)d_out;

    const int E = in_sizes[1] / 2;
    const int nblocks = (E + TILE_EDGES - 1) / TILE_EDGES;   // 15625 for E=2M
    mlp_decoder_wmma<<<nblocks, BLOCK, 0, stream>>>(z, ei, W1, b1, W2, b2, out, E);
}
